// MultiModalGC_69518340653373
// MI455X (gfx1250) — compile-verified, matrix operand order checked by reading, not run on hardware
//
#include <hip/hip_runtime.h>
#include <hip/hip_bf16.h>

// ---------------------------------------------------------------------------
// MultiModal graph conv pipeline for gfx1250 (CDNA5, wave32, WMMA bf16)
// ---------------------------------------------------------------------------

typedef __attribute__((ext_vector_type(16))) __bf16 v16bf;
typedef __attribute__((ext_vector_type(8)))  __bf16 bf16x8;
typedef __attribute__((ext_vector_type(8)))  float  v8f;

#define NNODES 50176      // 1024 * 49
#define NEDGES 86016      // 1024 * 84
#define HID    512
#define INDIM  1024
#define EB     336        // edge blocks (86016/256)
#define BNCH   98         // BN row chunks (50176/512)

// ---- fragment loader: lane L holds row/col (L&15), K-halves by (L>>4)*8 ----
__device__ inline v16bf load_frag16(const __bf16* __restrict__ p) {
    bf16x8 lo = *(const bf16x8*)(p);        // K = kb .. kb+7
    bf16x8 hi = *(const bf16x8*)(p + 16);   // K = 16+kb .. 16+kb+7
    return __builtin_shufflevector(lo, hi, 0,1,2,3,4,5,6,7,8,9,10,11,12,13,14,15);
}

// ---- closed-form grid-edge ids (per-graph, 84 edges) ----------------------
__device__ inline int edge_left (int r,int c){ return (c>0) ? ((r<6)?13*r+2*(c-1):78+(c-1)) : -1; }
__device__ inline int edge_right(int r,int c){ return (c<6) ? ((r<6)?13*r+2*c    :78+c    ) : -1; }
__device__ inline int edge_top  (int r,int c){ return (r>0) ? ((c<6)?13*(r-1)+2*c+1:13*(r-1)+12) : -1; }
__device__ inline int edge_down (int r,int c){ return (r<6) ? ((c<6)?13*r+2*c+1  :13*r+12 ) : -1; }

// ===========================================================================
// K0a: weight transpose + fp32 -> bf16  (WT[n][k] = W[k][n])
// ===========================================================================
__global__ __launch_bounds__(256)
void transpose_w(const float* __restrict__ W, __bf16* __restrict__ WT, int K, int Ncol) {
    int idx = blockIdx.x * 256 + threadIdx.x;
    if (idx >= K * Ncol) return;
    int k = idx % K, n = idx / K;
    WT[(size_t)n * K + k] = (__bf16)W[(size_t)k * Ncol + n];
}

// K0b: last row of dec_W (the node_w column weight)
__global__ __launch_bounds__(256)
void copy_dec_last(const float* __restrict__ decW, float* __restrict__ dec512) {
    int j = blockIdx.x * 256 + threadIdx.x;
    if (j < HID) dec512[j] = decW[(size_t)HID * HID + j];
}

// ===========================================================================
// K1: gather NCHW->node-major bf16 + fused coords = node @ proj_W + proj_b
// one block per node; L2 catches the 49-way hw reuse per (b,d) line
// ===========================================================================
__global__ __launch_bounds__(256)
void gather_nodes(const float* __restrict__ vis, const float* __restrict__ tac,
                  const float* __restrict__ projW, const float* __restrict__ projb,
                  __bf16* __restrict__ nodeB, float* __restrict__ coords) {
    __shared__ float s0[256], s1[256];
    int n = blockIdx.x;
    int b = n / 49, hw = n % 49;
    size_t base = (size_t)b * 512 * 49 + hw;
    float c0 = 0.f, c1 = 0.f;
    for (int d = threadIdx.x; d < INDIM; d += 256) {
        float v = (d < 512) ? vis[base + (size_t)d * 49]
                            : tac[base + (size_t)(d - 512) * 49];
        nodeB[(size_t)n * INDIM + d] = (__bf16)v;
        c0 += v * projW[2 * d];
        c1 += v * projW[2 * d + 1];
    }
    s0[threadIdx.x] = c0; s1[threadIdx.x] = c1;
    __syncthreads();
    for (int s = 128; s > 0; s >>= 1) {
        if (threadIdx.x < s) { s0[threadIdx.x] += s0[threadIdx.x + s];
                               s1[threadIdx.x] += s1[threadIdx.x + s]; }
        __syncthreads();
    }
    if (threadIdx.x == 0) {
        coords[2 * n]     = s0[0] + projb[0];
        coords[2 * n + 1] = s1[0] + projb[1];
    }
}

// ===========================================================================
// K2: per-edge attr = sigmoid(1/(dist+1e-6)); partial sums of exp(attr)
// ===========================================================================
__global__ __launch_bounds__(256)
void edge_kernel(const float* __restrict__ coords, float* __restrict__ edgeattr,
                 float* __restrict__ epart) {
    __shared__ float sh[256];
    int i = blockIdx.x * 256 + threadIdx.x;
    float al = 0.f;
    if (i < NEDGES) {
        int b = i / 84, e = i % 84;
        int r, c, down;
        if (e < 78) { r = e / 13; int m = e - r * 13;
                      if (m < 12) { c = m >> 1; down = m & 1; } else { c = 6; down = 1; } }
        else        { r = 6; c = e - 78; down = 0; }
        int src = b * 49 + r * 7 + c;
        int dst = src + (down ? 7 : 1);
        float dx = coords[2*src]   - coords[2*dst];
        float dy = coords[2*src+1] - coords[2*dst+1];
        float dist = sqrtf(dx * dx + dy * dy);
        float t  = 1.f / (dist + 1e-6f);
        float ea = 1.f / (1.f + expf(-t));
        edgeattr[i] = ea;
        al = expf(ea);
    }
    sh[threadIdx.x] = al; __syncthreads();
    for (int s = 128; s > 0; s >>= 1) {
        if (threadIdx.x < s) sh[threadIdx.x] += sh[threadIdx.x + s];
        __syncthreads();
    }
    if (threadIdx.x == 0) epart[blockIdx.x] = sh[0];
}

// K3: deterministic final softmax-denominator reduce -> 1/(sum+1e-8)
__global__ __launch_bounds__(256)
void edge_reduce(const float* __restrict__ epart, float* __restrict__ invsum) {
    __shared__ float sh[256];
    float s = 0.f;
    for (int i = threadIdx.x; i < EB; i += 256) s += epart[i];
    sh[threadIdx.x] = s; __syncthreads();
    for (int k = 128; k > 0; k >>= 1) {
        if (threadIdx.x < k) sh[threadIdx.x] += sh[threadIdx.x + k];
        __syncthreads();
    }
    if (threadIdx.x == 0) invsum[0] = 1.f / (sh[0] + 1e-8f);
}

// ===========================================================================
// K4: pull-based mean aggregation: agg[n] = (1/deg) * sum_in alpha_e * X[src]
// ===========================================================================
__global__ __launch_bounds__(256)
void aggregate(const __bf16* __restrict__ X, __bf16* __restrict__ AGG,
               const float* __restrict__ edgeattr, const float* __restrict__ invsum, int K) {
    int n = blockIdx.x;
    int b = n / 49, l = n % 49, r = l / 7, c = l % 7;
    int eL = edge_left(r, c), eT = edge_top(r, c);
    float inv = invsum[0];
    float deg = (float)((eL >= 0) + (eT >= 0));
    float ds  = 1.f / fmaxf(deg, 1.f);
    float wL = (eL >= 0) ? expf(edgeattr[b * 84 + eL]) * inv * ds : 0.f;
    float wT = (eT >= 0) ? expf(edgeattr[b * 84 + eT]) * inv * ds : 0.f;
    const __bf16* xL = X + (size_t)(n - 1) * K;
    const __bf16* xT = X + (size_t)(n - 7) * K;
    __bf16* out = AGG + (size_t)n * K;
    for (int d = threadIdx.x; d < K; d += 256) {
        float s = 0.f;
        if (eL >= 0) s += wL * (float)xL[d];
        if (eT >= 0) s += wT * (float)xT[d];
        out[d] = (__bf16)s;
    }
}

// ===========================================================================
// K5: dual WMMA GEMM  C = A1@W1T + A2@W2T + bias   (bf16 in, fp32 acc)
// block: 256 thr = 8 waves, 4(M)x2(N); block tile 128 x 128
// wave tile: 32 x 64  -> 8 accumulators, two A row-frags share each B frag
// MODE 0: store fp32 C[N][512]   MODE 1: +node_w rank-1, ReLU, NCHW store
// ===========================================================================
template<int MODE>
__global__ __launch_bounds__(256)
void gemm_wmma(const __bf16* __restrict__ A1, const __bf16* __restrict__ W1T,
               const __bf16* __restrict__ A2, const __bf16* __restrict__ W2T,
               const float* __restrict__ bias, int K,
               const float* __restrict__ nodew, const float* __restrict__ dec512,
               float* __restrict__ Cout) {
    const int lane = threadIdx.x & 31;
    const int wave = threadIdx.x >> 5;
    const int wm = wave & 3, wn = wave >> 2;
    const int rowBase = blockIdx.y * 128 + wm * 32;   // 32 rows per wave
    const int colBase = blockIdx.x * 128 + wn * 64;   // 64 cols per wave
    const int l15 = lane & 15;
    const int kb  = (lane >> 4) * 8;

    v8f acc[8];                                        // [t] rows 0-15, [4+t] rows 16-31
    #pragma unroll
    for (int t = 0; t < 8; ++t)
        #pragma unroll
        for (int i = 0; i < 8; ++i) acc[t][i] = 0.0f;

    #pragma unroll
    for (int pass = 0; pass < 2; ++pass) {
        const __bf16* A  = pass ? A2 : A1;
        const __bf16* WT = pass ? W2T : W1T;
        if (A == nullptr) continue;
        const __bf16* pa0 = A + (size_t)(rowBase + l15)      * K + kb;
        const __bf16* pa1 = A + (size_t)(rowBase + 16 + l15) * K + kb;
        for (int k0 = 0; k0 < K; k0 += 32) {
            v16bf a0 = load_frag16(pa0 + k0);
            v16bf a1 = load_frag16(pa1 + k0);
            #pragma unroll
            for (int t = 0; t < 4; ++t) {
                int col = colBase + t * 16 + l15;
                v16bf bf = load_frag16(WT + (size_t)col * K + k0 + kb);
                acc[t]     = __builtin_amdgcn_wmma_f32_16x16x32_bf16(
                    false, a0, false, bf, (short)0, acc[t],     false, false);
                acc[4 + t] = __builtin_amdgcn_wmma_f32_16x16x32_bf16(
                    false, a1, false, bf, (short)0, acc[4 + t], false, false);
            }
        }
    }

    // epilogue: lane L, vgpr v -> row = rg*16 + 8*(L>>4) + v, col = tile + (L&15)
    const int rt = 8 * (lane >> 4);
    #pragma unroll
    for (int t = 0; t < 4; ++t) {
        int col = colBase + t * 16 + l15;
        float bcol = bias[col];
        float d512 = (MODE == 1) ? dec512[col] : 0.f;
        #pragma unroll
        for (int rg = 0; rg < 2; ++rg) {
            int rtop = rowBase + rg * 16 + rt;
            #pragma unroll
            for (int v = 0; v < 8; ++v) {
                int row = rtop + v;
                float val = acc[rg * 4 + t][v] + bcol;
                if (MODE == 0) {
                    Cout[(size_t)row * HID + col] = val;
                } else {
                    val += nodew[row] * d512;
                    val = fmaxf(val, 0.f);
                    int b = row / 49, hw = row % 49;       // NCHW output
                    Cout[(size_t)b * (HID * 49) + (size_t)col * 49 + hw] = val;
                }
            }
        }
    }
}

// ===========================================================================
// K6: BN statistics (deterministic two-stage) + apply(BN+ReLU)->bf16
// ===========================================================================
__global__ __launch_bounds__(256)
void bn_stats1(const float* __restrict__ C, float* __restrict__ psum, float* __restrict__ psq) {
    int chunk = blockIdx.x;              // 98 chunks of 512 rows
    int r0 = chunk * 512;
    int c0 = threadIdx.x, c1 = threadIdx.x + 256;
    float s0 = 0, s1 = 0, q0 = 0, q1 = 0;
    for (int r = r0; r < r0 + 512; ++r) {
        float v0 = C[(size_t)r * HID + c0]; s0 += v0; q0 += v0 * v0;
        float v1 = C[(size_t)r * HID + c1]; s1 += v1; q1 += v1 * v1;
    }
    psum[chunk * HID + c0] = s0; psum[chunk * HID + c1] = s1;
    psq [chunk * HID + c0] = q0; psq [chunk * HID + c1] = q1;
}

__global__ __launch_bounds__(256)
void bn_stats2(const float* __restrict__ psum, const float* __restrict__ psq,
               const float* __restrict__ gamma, const float* __restrict__ beta,
               float* __restrict__ scale, float* __restrict__ shift) {
    int col = blockIdx.x * 256 + threadIdx.x;
    if (col >= HID) return;
    float s = 0, q = 0;
    for (int i = 0; i < BNCH; ++i) { s += psum[i * HID + col]; q += psq[i * HID + col]; }
    const float invN = 1.0f / (float)NNODES;
    float mu  = s * invN;
    float var = q * invN - mu * mu;
    float sc  = gamma[col] * rsqrtf(var + 1e-5f);
    scale[col] = sc;
    shift[col] = beta[col] - mu * sc;
}

__global__ __launch_bounds__(256)
void bn_apply(const float* __restrict__ C, const float* __restrict__ scale,
              const float* __restrict__ shift, __bf16* __restrict__ X) {
    size_t i = (size_t)blockIdx.x * 256 + threadIdx.x;
    int col = (int)(i & (HID - 1));
    float v = C[i] * scale[col] + shift[col];
    X[i] = (__bf16)fmaxf(v, 0.f);
}

// ===========================================================================
// K7: node_w = scatter_mean(edge_attr over both endpoints)  (pull, ≤4 edges)
// ===========================================================================
__global__ __launch_bounds__(256)
void node_weight(const float* __restrict__ edgeattr, float* __restrict__ nodew) {
    int n = blockIdx.x * 256 + threadIdx.x;
    if (n >= NNODES) return;
    int b = n / 49, l = n % 49, r = l / 7, c = l % 7;
    const float* ea = edgeattr + b * 84;
    float s = 0.f; int cnt = 0;
    int e;
    e = edge_left (r, c); if (e >= 0) { s += ea[e]; ++cnt; }
    e = edge_right(r, c); if (e >= 0) { s += ea[e]; ++cnt; }
    e = edge_top  (r, c); if (e >= 0) { s += ea[e]; ++cnt; }
    e = edge_down (r, c); if (e >= 0) { s += ea[e]; ++cnt; }
    nodew[n] = s / fmaxf((float)cnt, 1.f);
}

// ===========================================================================
// Host orchestration
// ===========================================================================
extern "C" void kernel_launch(void* const* d_in, const int* in_sizes, int n_in,
                              void* d_out, int out_size, void* d_ws, size_t ws_size,
                              hipStream_t stream) {
    (void)in_sizes; (void)n_in; (void)out_size; (void)ws_size;
    const float* vis      = (const float*)d_in[0];
    const float* tac      = (const float*)d_in[1];
    const float* projW    = (const float*)d_in[2];
    const float* projb    = (const float*)d_in[3];
    const float* gc1_relW = (const float*)d_in[4];
    const float* gc1_relb = (const float*)d_in[5];
    const float* gc1_rooW = (const float*)d_in[6];
    const float* bn1_g    = (const float*)d_in[7];
    const float* bn1_b    = (const float*)d_in[8];
    const float* gc2_relW = (const float*)d_in[9];
    const float* gc2_relb = (const float*)d_in[10];
    const float* gc2_rooW = (const float*)d_in[11];
    const float* bn2_g    = (const float*)d_in[12];
    const float* bn2_b    = (const float*)d_in[13];
    const float* decW     = (const float*)d_in[14];
    const float* decb     = (const float*)d_in[15];
    float* out = (float*)d_out;

    // ---- workspace carve-out (256B aligned) ----
    char* w = (char*)d_ws;
    size_t o = 0;
    auto alloc = [&](size_t bytes) -> char* {
        char* p = w + o; o += (bytes + 255) & ~(size_t)255; return p;
    };
    __bf16* WT1_rel = (__bf16*)alloc((size_t)HID * INDIM * 2);
    __bf16* WT1_roo = (__bf16*)alloc((size_t)HID * INDIM * 2);
    __bf16* WT2_rel = (__bf16*)alloc((size_t)HID * HID * 2);
    __bf16* WT2_roo = (__bf16*)alloc((size_t)HID * HID * 2);
    __bf16* WTdec   = (__bf16*)alloc((size_t)HID * HID * 2);
    float*  dec512  = (float*) alloc(HID * 4);
    float*  coords  = (float*) alloc((size_t)NNODES * 2 * 4);
    float*  edgeatt = (float*) alloc((size_t)NEDGES * 4);
    float*  epart   = (float*) alloc(EB * 4);
    float*  invsum  = (float*) alloc(4);
    float*  psum    = (float*) alloc((size_t)BNCH * HID * 4);
    float*  psq     = (float*) alloc((size_t)BNCH * HID * 4);
    float*  bnscale = (float*) alloc(HID * 4);
    float*  bnshift = (float*) alloc(HID * 4);
    float*  nodew   = (float*) alloc((size_t)NNODES * 4);
    __bf16* bufA    = (__bf16*)alloc((size_t)NNODES * INDIM * 2); // node, later X1|X2
    __bf16* bufB    = (__bf16*)alloc((size_t)NNODES * INDIM * 2); // agg1, later agg2
    float*  bufC    = (float*) alloc((size_t)NNODES * HID * 4);   // pre-BN activations

    __bf16* X1 = bufA;                           // overwrites node (after gemm1)
    __bf16* X2 = bufA + (size_t)NNODES * HID;    // second half

    // ---- K0: weight prep ----
    {
        int n1 = INDIM * HID, n2 = HID * HID;
        transpose_w<<<(n1 + 255) / 256, 256, 0, stream>>>(gc1_relW, WT1_rel, INDIM, HID);
        transpose_w<<<(n1 + 255) / 256, 256, 0, stream>>>(gc1_rooW, WT1_roo, INDIM, HID);
        transpose_w<<<(n2 + 255) / 256, 256, 0, stream>>>(gc2_relW, WT2_rel, HID, HID);
        transpose_w<<<(n2 + 255) / 256, 256, 0, stream>>>(gc2_rooW, WT2_roo, HID, HID);
        transpose_w<<<(n2 + 255) / 256, 256, 0, stream>>>(decW,     WTdec,   HID, HID);
        copy_dec_last<<<2, 256, 0, stream>>>(decW, dec512);
    }

    // ---- K1: gather + coords ----
    gather_nodes<<<NNODES, 256, 0, stream>>>(vis, tac, projW, projb, bufA, coords);

    // ---- K2/K3: edges + global softmax denominator ----
    edge_kernel<<<EB, 256, 0, stream>>>(coords, edgeatt, epart);
    edge_reduce<<<1, 256, 0, stream>>>(epart, invsum);

    // ---- K7: node weights (independent of GEMMs) ----
    node_weight<<<(NNODES + 255) / 256, 256, 0, stream>>>(edgeatt, nodew);

    dim3 gGrid(HID / 128, NNODES / 128);   // (4, 392)
    int  bnApplyBlocks = (NNODES * HID) / 256;

    // ---- layer 1 ----
    aggregate<<<NNODES, 256, 0, stream>>>(bufA, bufB, edgeatt, invsum, INDIM);
    gemm_wmma<0><<<gGrid, 256, 0, stream>>>(bufB, WT1_rel, bufA, WT1_roo,
                                            gc1_relb, INDIM, nullptr, nullptr, bufC);
    bn_stats1<<<BNCH, 256, 0, stream>>>(bufC, psum, psq);
    bn_stats2<<<2, 256, 0, stream>>>(psum, psq, bn1_g, bn1_b, bnscale, bnshift);
    bn_apply<<<bnApplyBlocks, 256, 0, stream>>>(bufC, bnscale, bnshift, X1);

    // ---- layer 2 ----
    aggregate<<<NNODES, 256, 0, stream>>>(X1, bufB, edgeatt, invsum, HID);
    gemm_wmma<0><<<gGrid, 256, 0, stream>>>(bufB, WT2_rel, X1, WT2_roo,
                                            gc2_relb, HID, nullptr, nullptr, bufC);
    bn_stats1<<<BNCH, 256, 0, stream>>>(bufC, psum, psq);
    bn_stats2<<<2, 256, 0, stream>>>(psum, psq, bn2_g, bn2_b, bnscale, bnshift);
    bn_apply<<<bnApplyBlocks, 256, 0, stream>>>(bufC, bnscale, bnshift, X2);

    // ---- decoder: GEMM + rank-1(node_w) + ReLU + NCHW store ----
    gemm_wmma<1><<<gGrid, 256, 0, stream>>>(X2, WTdec, nullptr, nullptr,
                                            decb, HID, nodew, dec512, out);
}
